// PeepholeLSTM_65240553226498
// MI455X (gfx1250) — matrix operand order repaired and checked by reading, not compiled, not run on hardware
//
#include <hip/hip_runtime.h>

// Problem dims (match reference)
#define B_  64
#define T_  512
#define D_  512
#define H_  1024
#define L_  2
#define G4H (4 * H_)

typedef __bf16 bf16_t;
typedef __attribute__((ext_vector_type(16))) __bf16 v16bf;
typedef __attribute__((ext_vector_type(8)))  __bf16 v8bf;
typedef __attribute__((ext_vector_type(8)))  float   v8f;

// ---- helpers -------------------------------------------------------------

__device__ __forceinline__ bf16_t f2bf(float f) {
  // round-to-nearest-even f32 -> bf16
  unsigned u = __float_as_uint(f);
  u += 0x7FFFu + ((u >> 16) & 1u);
  unsigned short s = (unsigned short)(u >> 16);
  return __builtin_bit_cast(bf16_t, s);
}

__device__ __forceinline__ v16bf cat8(v8bf lo, v8bf hi) {
  return __builtin_shufflevector(lo, hi, 0, 1, 2, 3, 4, 5, 6, 7,
                                         8, 9, 10, 11, 12, 13, 14, 15);
}

// A block (16x32 bf16, MxK). base already includes row + half*16 offset.
// lane<16: K = k..k+15 ; lane>=16: K = k+16..k+31  (contiguous in memory)
__device__ __forceinline__ v16bf load_a(const bf16_t* __restrict__ base, int k) {
  v8bf lo = *(const v8bf*)(base + k);
  v8bf hi = *(const v8bf*)(base + k + 8);
  return cat8(lo, hi);
}

// B block (32x16 bf16, KxN). rowbase = W row (column of B) + half*16 offset.
// lane<16: K = k..k+15 ; lane>=16: K = k+16..k+31  (contiguous in memory)
__device__ __forceinline__ v16bf load_b(const bf16_t* __restrict__ rowbase, int k) {
  v8bf lo = *(const v8bf*)(rowbase + k);
  v8bf hi = *(const v8bf*)(rowbase + k + 8);
  return cat8(lo, hi);
}

__device__ __forceinline__ float sigmoidf_(float x) {
  return 1.0f / (1.0f + __expf(-x));
}

// ---- elementwise utility kernels ----------------------------------------

__global__ void cvt_f32_to_bf16(const float* __restrict__ src,
                                bf16_t* __restrict__ dst, size_t n) {
  size_t i = (size_t)blockIdx.x * blockDim.x + threadIdx.x;
  size_t s = (size_t)gridDim.x * blockDim.x;
  for (; i < n; i += s) dst[i] = f2bf(src[i]);
}

__global__ void zero_u32(unsigned* __restrict__ p, size_t n) {
  size_t i = (size_t)blockIdx.x * blockDim.x + threadIdx.x;
  size_t s = (size_t)gridDim.x * blockDim.x;
  for (; i < n; i += s) p[i] = 0u;
}

// ---- fused LSTM step (one layer, one timestep) ---------------------------
// grid = 32 blocks x 256 threads = 256 waves; each wave computes a 16x16
// output tile for ALL FOUR gates (m-tile of B, n-tile of H), then applies the
// peephole gate math in-register.
__global__ __launch_bounds__(256)
void PeepholeLSTM_step_kernel(
    const bf16_t* __restrict__ Ain, long strideA, int Kin,  // input act [B, Kin]
    const bf16_t* __restrict__ Wx,                          // [4H, Kin]
    const bf16_t* __restrict__ hprev,                       // [B, H] bf16
    const bf16_t* __restrict__ Wh,                          // [4H, H]
    const float* __restrict__ bx,                           // [4H]
    const float* __restrict__ bh,                           // [4H]
    const float* __restrict__ peep,                         // [3, H]
    float* __restrict__ c_state,                            // [B, H] f32 (in/out)
    float* __restrict__ h_state,                            // [B, H] f32 (out, = h_n slot)
    bf16_t* __restrict__ hnext,                             // [B, H] bf16 (out)
    float* __restrict__ y, long strideY)                    // optional [B, T, H] slice
{
  const int lane = threadIdx.x & 31;
  const int wid  = blockIdx.x * (blockDim.x >> 5) + (threadIdx.x >> 5);
  const int m0   = (wid & 3) << 4;        // 64/16 = 4 m-tiles
  const int n0   = (wid >> 2) << 4;       // 1024/16 = 64 n-tiles
  const int lrow = lane & 15;
  const int half = lane >> 4;

  v8f acc0 = {}, acc1 = {}, acc2 = {}, acc3 = {};

  // ---- GEMM 1: input projection  z += A @ Wx^T ----
  {
    const bf16_t* abase = Ain + (size_t)(m0 + lrow) * (size_t)strideA + half * 16;
    const bf16_t* b0 = Wx + (size_t)(0 * H_ + n0 + lrow) * Kin + half * 16;
    const bf16_t* b1 = Wx + (size_t)(1 * H_ + n0 + lrow) * Kin + half * 16;
    const bf16_t* b2 = Wx + (size_t)(2 * H_ + n0 + lrow) * Kin + half * 16;
    const bf16_t* b3 = Wx + (size_t)(3 * H_ + n0 + lrow) * Kin + half * 16;
    for (int k = 0; k < Kin; k += 32) {
      v16bf a = load_a(abase, k);
      acc0 = __builtin_amdgcn_wmma_f32_16x16x32_bf16(false, a, false, load_b(b0, k), (short)0, acc0, false, false);
      acc1 = __builtin_amdgcn_wmma_f32_16x16x32_bf16(false, a, false, load_b(b1, k), (short)0, acc1, false, false);
      acc2 = __builtin_amdgcn_wmma_f32_16x16x32_bf16(false, a, false, load_b(b2, k), (short)0, acc2, false, false);
      acc3 = __builtin_amdgcn_wmma_f32_16x16x32_bf16(false, a, false, load_b(b3, k), (short)0, acc3, false, false);
    }
  }

  // ---- GEMM 2: recurrent projection  z += h_prev @ Wh^T ----
  {
    const bf16_t* hbase = hprev + (size_t)(m0 + lrow) * H_ + half * 16;
    const bf16_t* b0 = Wh + (size_t)(0 * H_ + n0 + lrow) * H_ + half * 16;
    const bf16_t* b1 = Wh + (size_t)(1 * H_ + n0 + lrow) * H_ + half * 16;
    const bf16_t* b2 = Wh + (size_t)(2 * H_ + n0 + lrow) * H_ + half * 16;
    const bf16_t* b3 = Wh + (size_t)(3 * H_ + n0 + lrow) * H_ + half * 16;
    for (int k = 0; k < H_; k += 32) {
      v16bf a = load_a(hbase, k);
      acc0 = __builtin_amdgcn_wmma_f32_16x16x32_bf16(false, a, false, load_b(b0, k), (short)0, acc0, false, false);
      acc1 = __builtin_amdgcn_wmma_f32_16x16x32_bf16(false, a, false, load_b(b1, k), (short)0, acc1, false, false);
      acc2 = __builtin_amdgcn_wmma_f32_16x16x32_bf16(false, a, false, load_b(b2, k), (short)0, acc2, false, false);
      acc3 = __builtin_amdgcn_wmma_f32_16x16x32_bf16(false, a, false, load_b(b3, k), (short)0, acc3, false, false);
    }
  }

  // ---- gate epilogue (C/D layout: lane holds col n0+lrow, rows 8*half+r) --
  const int n = n0 + lrow;
  const float bi = bx[0 * H_ + n] + bh[0 * H_ + n];
  const float bf = bx[1 * H_ + n] + bh[1 * H_ + n];
  const float bg = bx[2 * H_ + n] + bh[2 * H_ + n];
  const float bo = bx[3 * H_ + n] + bh[3 * H_ + n];
  const float pi = peep[0 * H_ + n];
  const float pf = peep[1 * H_ + n];
  const float po = peep[2 * H_ + n];
  const int mbase = m0 + half * 8;

#pragma unroll
  for (int r = 0; r < 8; ++r) {
    const int m = mbase + r;
    const size_t idx = (size_t)m * H_ + n;
    const float c_old = c_state[idx];
    const float zi = acc0[r] + bi;
    const float zf = acc1[r] + bf;
    const float zg = acc2[r] + bg;
    const float zo = acc3[r] + bo;
    const float ig = sigmoidf_(zi + pi * c_old);
    const float fg = sigmoidf_(zf + pf * c_old);
    const float gg = tanhf(zg);
    const float cn = fg * c_old + ig * gg;
    const float og = sigmoidf_(zo + po * cn);
    const float hn = og * tanhf(cn);
    c_state[idx] = cn;
    h_state[idx] = hn;
    hnext[idx]   = f2bf(hn);
    if (y) y[(size_t)m * (size_t)strideY + n] = hn;
  }
}

// ---- host-side orchestration --------------------------------------------

extern "C" void kernel_launch(void* const* d_in, const int* in_sizes, int n_in,
                              void* d_out, int out_size, void* d_ws, size_t ws_size,
                              hipStream_t stream) {
  (void)in_sizes; (void)n_in; (void)out_size; (void)ws_size;

  const float* x    = (const float*)d_in[0];  // [B, T, D]
  const float* Wx0  = (const float*)d_in[1];  // [4H, D]
  const float* bx0  = (const float*)d_in[2];  // [4H]
  const float* Wx1  = (const float*)d_in[3];  // [4H, H]
  const float* bx1  = (const float*)d_in[4];  // [4H]
  const float* Wh   = (const float*)d_in[5];  // [L, 4H, H]
  const float* bh   = (const float*)d_in[6];  // [L, 4H]
  const float* peep = (const float*)d_in[7];  // [L, 3, H]

  float* out = (float*)d_out;
  float* y   = out;                               // [B, T, H]
  float* h_n = out + (size_t)B_ * T_ * H_;        // [L, B, H]
  float* c_n = h_n + (size_t)L_ * B_ * H_;        // [L, B, H]

  // workspace layout (bf16)
  bf16_t* xb   = (bf16_t*)d_ws;                       // [B, T, D]
  bf16_t* wx0b = xb   + (size_t)B_ * T_ * D_;         // [4H, D]
  bf16_t* wx1b = wx0b + (size_t)G4H * D_;             // [4H, H]
  bf16_t* whb  = wx1b + (size_t)G4H * H_;             // [L, 4H, H]
  bf16_t* h0b  = whb  + (size_t)L_ * G4H * H_;        // 2 x [B, H] (double buffer)
  bf16_t* h1b  = h0b  + 2 * (size_t)B_ * H_;          // 2 x [B, H]

  // one-time conversions (per call; deterministic)
  cvt_f32_to_bf16<<<2048, 256, 0, stream>>>(x,   xb,   (size_t)B_ * T_ * D_);
  cvt_f32_to_bf16<<<512,  256, 0, stream>>>(Wx0, wx0b, (size_t)G4H * D_);
  cvt_f32_to_bf16<<<512,  256, 0, stream>>>(Wx1, wx1b, (size_t)G4H * H_);
  cvt_f32_to_bf16<<<1024, 256, 0, stream>>>(Wh,  whb,  (size_t)L_ * G4H * H_);

  // zero c-state (read at t=0) and bf16 h double-buffers (buf0 read at t=0)
  zero_u32<<<256, 256, 0, stream>>>((unsigned*)c_n, (size_t)L_ * B_ * H_);
  zero_u32<<<256, 256, 0, stream>>>((unsigned*)h0b, (size_t)4 * B_ * H_ / 2);

  const dim3 grd(32), blk(256);
  const size_t BH = (size_t)B_ * H_;

  for (int t = 0; t < T_; ++t) {
    const bf16_t* h0prev = h0b + (size_t)(t & 1) * BH;
    bf16_t*       h0next = h0b + (size_t)((t + 1) & 1) * BH;
    const bf16_t* h1prev = h1b + (size_t)(t & 1) * BH;
    bf16_t*       h1next = h1b + (size_t)((t + 1) & 1) * BH;

    // layer 0: A = x[:, t, :]  (row stride T*D), K = D
    PeepholeLSTM_step_kernel<<<grd, blk, 0, stream>>>(
        xb + (size_t)t * D_, (long)T_ * D_, D_,
        wx0b, h0prev, whb,
        bx0, bh, peep,
        c_n, h_n, h0next,
        (float*)nullptr, 0L);

    // layer 1: A = h0(t)  (row stride H), K = H
    PeepholeLSTM_step_kernel<<<grd, blk, 0, stream>>>(
        h0next, (long)H_, H_,
        wx1b, h1prev, whb + (size_t)G4H * H_,
        bx1, bh + G4H, peep + 3 * H_,
        c_n + BH, h_n + BH, h1next,
        y + (size_t)t * H_, (long)T_ * H_);
  }
}
